// HyperbolicAttention_4123168604893
// MI455X (gfx1250) — compile-verified, matrix operand order checked by reading, not run on hardware
//
#include <hip/hip_runtime.h>
#include <hip/hip_bf16.h>

typedef __attribute__((ext_vector_type(16))) _Float16 v16h;
typedef __attribute__((ext_vector_type(8)))  float    v8f;

#define EPS 1e-7f
#define NEG_BIG (-1e30f)

static constexpr int N = 512, D = 512, H = 8, HD = 64, NOUT3 = 1536;

// ---------------------------------------------------------------------------
// WMMA helpers (CDNA5: V_WMMA_F32_16X16X32_F16, wave32)
// ---------------------------------------------------------------------------
__device__ __forceinline__ v8f wmma_f16(v16h a, v16h b, v8f c) {
  // 8 args: (neg_a, A, neg_b, B, c_mod, C, reuse_a, reuse_b)
  return __builtin_amdgcn_wmma_f32_16x16x32_f16(false, a, false, b, (short)0, c,
                                                false, false);
}

// Load a 16x32 f16 A-fragment (or B-fragment from transposed storage) from a
// row-major matrix with leading dimension ld, starting at (row0, k0).
// ISA layout: lane l holds row (l&15); half-group (l>>4) selects K sub-blocks:
//   regs 0..7  -> K = half*8 + 0..7
//   regs 8..15 -> K = 16 + half*8 + 0..7
__device__ __forceinline__ v16h load_frag16(const _Float16* p, int ld, int row0,
                                            int k0, int lane) {
  int m  = lane & 15;
  int hv = (lane >> 4) & 1;
  const _Float16* rp = p + (row0 + m) * ld + k0 + hv * 8;
  v16h a;
#pragma unroll
  for (int j = 0; j < 8; ++j) a[j] = rp[j];
#pragma unroll
  for (int j = 0; j < 8; ++j) a[8 + j] = rp[16 + j];
  return a;
}

// ---------------------------------------------------------------------------
// Kernel 1: convert + transpose weights to f16 (B-operand layout)
// ---------------------------------------------------------------------------
__global__ void prep_weights(const float* __restrict__ w_qkv,
                             const float* __restrict__ w_out,
                             _Float16* __restrict__ wqkvT,
                             _Float16* __restrict__ woutT) {
  int idx = blockIdx.x * blockDim.x + threadIdx.x;
  const int e1 = NOUT3 * D;
  const int e2 = D * D;
  if (idx < e1) {
    int no = idx / D, k = idx - no * D;
    wqkvT[idx] = (_Float16)w_qkv[k * NOUT3 + no];
  } else if (idx < e1 + e2) {
    int j = idx - e1;
    int no = j / D, k = j - no * D;
    woutT[j] = (_Float16)w_out[k * D + no];
  }
}

// ---------------------------------------------------------------------------
// Kernel 2: x_tan = logmap0(x_hyp, c_sphere), stored f16 for the QKV GEMM
// ---------------------------------------------------------------------------
__global__ __launch_bounds__(256) void logmap_kernel(
    const float* __restrict__ x, const float* __restrict__ c_sphere,
    _Float16* __restrict__ x_tan) {
  int row = blockIdx.x;
  int tid = threadIdx.x;
  __shared__ float red[256];
  float v0 = x[row * D + tid];
  float v1 = x[row * D + tid + 256];
  red[tid] = v0 * v0 + v1 * v1;
  __syncthreads();
#pragma unroll
  for (int s = 128; s > 0; s >>= 1) {
    if (tid < s) red[tid] += red[tid + s];
    __syncthreads();
  }
  float yn  = sqrtf(red[0]);
  float sqc = fmaxf(sqrtf(c_sphere[0]), EPS);
  float z   = fminf(yn, 1.0f - EPS);
  float mag = 0.5f * logf((1.0f + z) / (1.0f - z)) / sqc;  // arctanh(z)/sqrt(c)
  float scl = (yn < EPS) ? 0.0f : mag / fmaxf(yn, EPS);
  x_tan[row * D + tid]       = (_Float16)(v0 * scl);
  x_tan[row * D + tid + 256] = (_Float16)(v1 * scl);
}

// ---------------------------------------------------------------------------
// Kernel 3/6: C(f32) = A(f16, MxK) * B(from BT f16, KxN) + bias ; one wave per
// 16x16 output tile, K-loop in steps of 32.
// ---------------------------------------------------------------------------
__global__ __launch_bounds__(32) void wmma_gemm_bias(
    const _Float16* __restrict__ A, const _Float16* __restrict__ BT,
    const float* __restrict__ bias, float* __restrict__ Cout, int K, int ldc) {
  int n0 = blockIdx.x * 16;
  int m0 = blockIdx.y * 16;
  int lane = threadIdx.x;
  int nn = lane & 15, hv = lane >> 4;
  v8f acc = {};
  for (int k0 = 0; k0 < K; k0 += 32) {
    v16h a = load_frag16(A, K, m0, k0, lane);
    v16h b = load_frag16(BT, K, n0, k0, lane);
    acc = wmma_f16(a, b, acc);
  }
  float bv = bias[n0 + nn];
#pragma unroll
  for (int r = 0; r < 8; ++r)
    Cout[(m0 + r + 8 * hv) * ldc + n0 + nn] = acc[r] + bv;
}

// ---------------------------------------------------------------------------
// Kernel 4: RoPE + expmap0 for q/k; also ||.||^2 per (h,n) and V transposed.
// One 64-thread block per (token n, head h).
// ---------------------------------------------------------------------------
__global__ __launch_bounds__(64) void rope_exp_kernel(
    const float* __restrict__ qkv, const float* __restrict__ freqs,
    const float* __restrict__ c_logits, _Float16* __restrict__ q_hyp,
    _Float16* __restrict__ k_hyp, _Float16* __restrict__ vt,
    float* __restrict__ qq, float* __restrict__ kk) {
  int n = blockIdx.x, h = blockIdx.y, d = threadIdx.x;
  __shared__ float sq[64], sk[64];
  const float* base = qkv + n * NOUT3 + h * HD;
  float qd = base[d],     qp = base[d ^ 32];
  float kd = base[D + d], kp = base[D + (d ^ 32)];
  float vv = base[2 * D + d];
  int j = d & 31;
  float ang = freqs[n * 32 + j];
  float cs = cosf(ang), sn = sinf(ang);
  // rotated = concat(re, im); re[j] = xr*cos - xi*sin ; im[j] = xr*sin + xi*cos
  float qr = (d < 32) ? (qd * cs - qp * sn) : (qp * sn + qd * cs);
  float kr = (d < 32) ? (kd * cs - kp * sn) : (kp * sn + kd * cs);
  sq[d] = qr * qr;
  sk[d] = kr * kr;
  __syncthreads();
#pragma unroll
  for (int s = 32; s > 0; s >>= 1) {
    if (d < s) { sq[d] += sq[d + s]; sk[d] += sk[d + s]; }
    __syncthreads();
  }
  float c   = log1pf(expf(c_logits[h]));     // softplus
  float sqc = fmaxf(sqrtf(c), EPS);
  float qn = sqrtf(sq[0]), kn = sqrtf(sk[0]);
  float qmag = tanhf(sqc * qn) / sqc; if (qmag >= 1.0f) qmag = 1.0f - EPS;  // project
  float kmag = tanhf(sqc * kn) / sqc; if (kmag >= 1.0f) kmag = 1.0f - EPS;
  float qs = (qn < EPS) ? 0.0f : qmag / fmaxf(qn, EPS);
  float ks = (kn < EPS) ? 0.0f : kmag / fmaxf(kn, EPS);
  q_hyp[(h * N + n) * HD + d] = (_Float16)(qr * qs);
  k_hyp[(h * N + n) * HD + d] = (_Float16)(kr * ks);
  vt[(h * HD + d) * N + n]    = (_Float16)vv;
  if (d == 0) {
    qq[h * N + n] = (qn < EPS) ? 0.0f : qmag * qmag;
    kk[h * N + n] = (kn < EPS) ? 0.0f : kmag * kmag;
  }
}

// ---------------------------------------------------------------------------
// Hyperbolic score from dot product + squared norms (closed-form Mobius dist)
// ---------------------------------------------------------------------------
__device__ __forceinline__ float hyp_score(float t, float a2, float b2, float c,
                                           float sqc, float gs) {
  float alpha = 1.0f - 2.0f * c * t + c * b2;
  float beta  = 1.0f - c * a2;
  float num2  = alpha * alpha * a2 - 2.0f * alpha * beta * t + beta * beta * b2;
  float den   = fmaxf(1.0f - 2.0f * c * t + c * c * a2 * b2, EPS);
  float nrm   = sqrtf(fmaxf(num2, 0.0f)) / den;
  if (nrm >= 1.0f) nrm = 1.0f - EPS;                    // _project
  float arg  = fminf(sqc * nrm, 1.0f - EPS);
  float dist = logf((1.0f + arg) / (1.0f - arg)) / sqc;  // 2*arctanh(arg)/sqc
  return -gs * dist;
}

// ---------------------------------------------------------------------------
// Kernel 5: fused flash-style hyperbolic attention. One wave per (16-query
// tile, head). Key tiles of 32, causal, online softmax, P*V via WMMA.
// ---------------------------------------------------------------------------
__global__ __launch_bounds__(32) void attn_kernel(
    const _Float16* __restrict__ qh, const _Float16* __restrict__ kh,
    const _Float16* __restrict__ vt, const float* __restrict__ qq,
    const float* __restrict__ kk, const float* __restrict__ c_logits,
    const float* __restrict__ geo_scale, _Float16* __restrict__ attn_out) {
  int qt = blockIdx.x, h = blockIdx.y;
  int q0 = qt * 16;
  int lane = threadIdx.x;
  int nn = lane & 15, hv = lane >> 4;
  const _Float16* qh_h = qh + h * N * HD;
  const _Float16* kh_h = kh + h * N * HD;
  const _Float16* vt_h = vt + h * HD * N;
  const float* qq_h = qq + h * N;
  const float* kk_h = kk + h * N;
  float c   = log1pf(expf(c_logits[h]));
  float sqc = fmaxf(sqrtf(c), EPS);
  float gs  = geo_scale[h];

  float qqv[8];
#pragma unroll
  for (int r = 0; r < 8; ++r) qqv[r] = qq_h[q0 + r + 8 * hv];

  v16h aq0 = load_frag16(qh_h, HD, q0, 0, lane);
  v16h aq1 = load_frag16(qh_h, HD, q0, 32, lane);

  float M[8], L[8];
  v8f o0 = {}, o1 = {}, o2 = {}, o3 = {};
#pragma unroll
  for (int r = 0; r < 8; ++r) { M[r] = NEG_BIG; L[r] = 0.0f; }

  __shared__ _Float16 plds[16][32];

  int ntiles = (q0 + 15) / 32 + 1;  // causal: only key tiles with k0 <= q0+15
  for (int kt = 0; kt < ntiles; ++kt) {
    int k0 = kt * 32;
    // S = Q . K^T for a 16x32 score tile (two 16x16 WMMA outputs)
    v8f s1 = {}, s2 = {};
    {
      v16h b1 = load_frag16(kh_h, HD, k0, 0, lane);
      v16h b2 = load_frag16(kh_h, HD, k0 + 16, 0, lane);
      s1 = wmma_f16(aq0, b1, s1);
      s2 = wmma_f16(aq0, b2, s2);
      b1 = load_frag16(kh_h, HD, k0, 32, lane);
      b2 = load_frag16(kh_h, HD, k0 + 16, 32, lane);
      s1 = wmma_f16(aq1, b1, s1);
      s2 = wmma_f16(aq1, b2, s2);
    }
    float b2a = kk_h[k0 + nn];
    float b2b = kk_h[k0 + 16 + nn];
    int kj1 = k0 + nn, kj2 = k0 + 16 + nn;
    float p1[8], p2[8], tmax[8];
#pragma unroll
    for (int r = 0; r < 8; ++r) {
      int qi = q0 + r + 8 * hv;
      float sc1 = hyp_score(s1[r], qqv[r], b2a, c, sqc, gs);
      float sc2 = hyp_score(s2[r], qqv[r], b2b, c, sqc, gs);
      if (kj1 > qi) sc1 = NEG_BIG;  // causal mask
      if (kj2 > qi) sc2 = NEG_BIG;
      p1[r] = sc1;
      p2[r] = sc2;
      tmax[r] = fmaxf(sc1, sc2);
    }
    // row-wise max across the 16 lanes of this half-group (C-frag row stripe)
#pragma unroll
    for (int mxs = 1; mxs < 16; mxs <<= 1)
#pragma unroll
      for (int r = 0; r < 8; ++r)
        tmax[r] = fmaxf(tmax[r], __shfl_xor(tmax[r], mxs, 32));
    float scl[8], rsum[8];
#pragma unroll
    for (int r = 0; r < 8; ++r) {
      float mn = fmaxf(M[r], tmax[r]);
      scl[r] = expf(M[r] - mn);
      M[r] = mn;
      p1[r] = expf(p1[r] - mn);
      p2[r] = expf(p2[r] - mn);
      rsum[r] = p1[r] + p2[r];
    }
#pragma unroll
    for (int mxs = 1; mxs < 16; mxs <<= 1)
#pragma unroll
      for (int r = 0; r < 8; ++r)
        rsum[r] += __shfl_xor(rsum[r], mxs, 32);
#pragma unroll
    for (int r = 0; r < 8; ++r) {
      L[r] = L[r] * scl[r] + rsum[r];
      o0[r] *= scl[r]; o1[r] *= scl[r]; o2[r] *= scl[r]; o3[r] *= scl[r];
      plds[r + 8 * hv][nn]      = (_Float16)p1[r];   // C-layout -> LDS
      plds[r + 8 * hv][16 + nn] = (_Float16)p2[r];
    }
    __syncthreads();
    // P (16x32) as A-frag; V tile columns from Vt rows as B-frags
    v16h pA  = load_frag16(&plds[0][0], 32, 0, 0, lane);
    v16h bv0 = load_frag16(vt_h, N, 0, k0, lane);
    v16h bv1 = load_frag16(vt_h, N, 16, k0, lane);
    v16h bv2 = load_frag16(vt_h, N, 32, k0, lane);
    v16h bv3 = load_frag16(vt_h, N, 48, k0, lane);
    o0 = wmma_f16(pA, bv0, o0);
    o1 = wmma_f16(pA, bv1, o1);
    o2 = wmma_f16(pA, bv2, o2);
    o3 = wmma_f16(pA, bv3, o3);
    __syncthreads();
  }
#pragma unroll
  for (int r = 0; r < 8; ++r) {
    float inv = 1.0f / L[r];
    int row = q0 + r + 8 * hv;
    _Float16* op = attn_out + row * D + h * HD;
    op[nn]      = (_Float16)(o0[r] * inv);
    op[16 + nn] = (_Float16)(o1[r] * inv);
    op[32 + nn] = (_Float16)(o2[r] * inv);
    op[48 + nn] = (_Float16)(o3[r] * inv);
  }
}

// ---------------------------------------------------------------------------
extern "C" void kernel_launch(void* const* d_in, const int* in_sizes, int n_in,
                              void* d_out, int out_size, void* d_ws,
                              size_t ws_size, hipStream_t stream) {
  (void)in_sizes; (void)n_in; (void)out_size; (void)ws_size;
  const float* x_hyp     = (const float*)d_in[0];
  const float* freqs     = (const float*)d_in[1];
  const float* c_sphere  = (const float*)d_in[2];
  const float* w_qkv     = (const float*)d_in[3];
  const float* b_qkv     = (const float*)d_in[4];
  const float* w_out     = (const float*)d_in[5];
  const float* b_out     = (const float*)d_in[6];
  const float* c_logits  = (const float*)d_in[7];
  const float* geo_scale = (const float*)d_in[8];
  float* out = (float*)d_out;

  char* ws = (char*)d_ws;
  size_t off = 0;
  auto alloc = [&](size_t bytes) {
    char* p = ws + off;
    off = (off + bytes + 255) & ~(size_t)255;
    return p;
  };
  _Float16* x_tan  = (_Float16*)alloc((size_t)N * D * 2);
  _Float16* wqkvT  = (_Float16*)alloc((size_t)NOUT3 * D * 2);
  _Float16* woutT  = (_Float16*)alloc((size_t)D * D * 2);
  float*    qkv    = (float*)   alloc((size_t)N * NOUT3 * 4);
  _Float16* q_hyp  = (_Float16*)alloc((size_t)H * N * HD * 2);
  _Float16* k_hyp  = (_Float16*)alloc((size_t)H * N * HD * 2);
  _Float16* vt     = (_Float16*)alloc((size_t)H * HD * N * 2);
  float*    qq     = (float*)   alloc((size_t)H * N * 4);
  float*    kk     = (float*)   alloc((size_t)H * N * 4);
  _Float16* attn_o = (_Float16*)alloc((size_t)N * D * 2);

  int prep_elems = NOUT3 * D + D * D;
  prep_weights<<<(prep_elems + 255) / 256, 256, 0, stream>>>(w_qkv, w_out,
                                                             wqkvT, woutT);
  logmap_kernel<<<N, 256, 0, stream>>>(x_hyp, c_sphere, x_tan);
  wmma_gemm_bias<<<dim3(NOUT3 / 16, N / 16), 32, 0, stream>>>(
      x_tan, wqkvT, b_qkv, qkv, D, NOUT3);
  rope_exp_kernel<<<dim3(N, H), 64, 0, stream>>>(qkv, freqs, c_logits, q_hyp,
                                                 k_hyp, vt, qq, kk);
  attn_kernel<<<dim3(N / 16, H), 32, 0, stream>>>(q_hyp, k_hyp, vt, qq, kk,
                                                  c_logits, geo_scale, attn_o);
  wmma_gemm_bias<<<dim3(D / 16, N / 16), 32, 0, stream>>>(attn_o, woutT, b_out,
                                                          out, D, D);
}
